// EGAT_49890340110825
// MI455X (gfx1250) — matrix operand order repaired and checked by reading, not compiled, not run on hardware
//
#include <hip/hip_runtime.h>
#include <hip/hip_bf16.h>

// ---------------------------------------------------------------------------
// EGAT on gfx1250 (MI455X): f16 WMMA for all GEMMs (f32 accumulate),
// f32 softmax / ELU / log-softmax. HBM-bound (~1 GB attention traffic),
// so data movement is minimized: fused epilogues, in-place attention chain,
// B operands stored column-major so every WMMA fragment is 2x b128 loads.
// ---------------------------------------------------------------------------

typedef _Float16 f16;
typedef _Float16 v16h __attribute__((ext_vector_type(16)));
typedef _Float16 v8h  __attribute__((ext_vector_type(8)));
typedef float    v8f  __attribute__((ext_vector_type(8)));

#define NNODES  2048
#define NFEATC  256
#define PP      4
#define NHID    64
#define NHEADSC 4
#define NCLASSC 16
#define OUT_INC 1024   // NHID*NHEADS*P

// ---------------------------------------------------------------------------
// Generic batched WMMA GEMM: C[M,16*Ntiles] (f32 acc) = A[M,K] * B[K,16*Ntiles]
//   A: row-major f16, lda = K (+ batch stride)
//   B: COLUMN-major f16 (Bcm[n*ldb + k]), ldb = K (+ batch stride)
// Fragment layouts per CDNA5 ISA 7.12.2 (wave32):
//   A 16x32 f16: lane m = lane&15; K in [kb,kb+8) U [kb+16,kb+24), kb = lane<16?0:8
//   B 32x16 f16: lane n = lane&15; K in [kb,kb+8) U [kb+16,kb+24), kb = lane<16?0:8
//   C 16x16 f32: elem r -> M = r + (lane<16?0:8), N = lane&15
// Epilogue modes:
//   0: C32[m*ldc+n] = v  AND  C16[n*c16ld+m] = (f16)v   (col-major f16 copy
//      -> directly usable as the next GEMM's B operand)
//   1: ELU(v) -> C16[m*c16ld + colOffBase + batch*colOffPerBatch + n]
//   2: C32[m*ldc+n] = v  (batched)
// ---------------------------------------------------------------------------
__global__ __launch_bounds__(128) void egat_gemm_wmma(
    const f16* __restrict__ A, int lda, size_t aBatch,
    const f16* __restrict__ Bcm, int ldb, size_t bBatch,
    float* __restrict__ C32, int ldc, size_t cBatch,
    f16* __restrict__ C16, int c16ld, size_t c16Batch,
    int colOffBase, int colOffPerBatch,
    int M, int Ntiles, int K, int mode)
{
  const int wave = threadIdx.x >> 5;
  const int lane = threadIdx.x & 31;
  const int tile = blockIdx.x * 4 + wave;
  const int mt   = tile / Ntiles;
  const int nt   = tile - mt * Ntiles;
  const int b    = blockIdx.z;
  if (mt * 16 >= M) return;

  const int l15 = lane & 15;
  const int kbA = (lane < 16) ? 0 : 8;
  const int kbB = (lane < 16) ? 0 : 8;
  const f16* Arow = A   + b * aBatch + (size_t)(mt * 16 + l15) * (size_t)lda;
  const f16* Bcol = Bcm + b * bBatch + (size_t)(nt * 16 + l15) * (size_t)ldb;

  v8f acc = {};
  for (int k0 = 0; k0 < K; k0 += 32) {
    v8h alo = *(const v8h*)(Arow + k0 + kbA);
    v8h ahi = *(const v8h*)(Arow + k0 + kbA + 16);
    v8h blo = *(const v8h*)(Bcol + k0 + kbB);
    v8h bhi = *(const v8h*)(Bcol + k0 + kbB + 16);
    v16h av, bv;
#pragma unroll
    for (int t = 0; t < 8; ++t) {
      av[t] = alo[t]; av[t + 8] = ahi[t];
      bv[t] = blo[t]; bv[t + 8] = bhi[t];
    }
    acc = __builtin_amdgcn_wmma_f32_16x16x32_f16(
        /*neg_a=*/false, av, /*neg_b=*/false, bv,
        /*c_mod=*/(short)0, acc, /*reuse_a=*/false, /*reuse_b=*/false);
  }

  const int n  = nt * 16 + l15;
  const int m0 = mt * 16 + ((lane < 16) ? 0 : 8);
  if (mode == 0) {
#pragma unroll
    for (int r = 0; r < 8; ++r) {
      const int m = m0 + r;
      const float v = acc[r];
      C32[b * cBatch + (size_t)m * ldc + n]      = v;
      C16[b * c16Batch + (size_t)n * c16ld + m] = (f16)v;
    }
  } else if (mode == 1) {
    const int colOff = colOffBase + b * colOffPerBatch;
#pragma unroll
    for (int r = 0; r < 8; ++r) {
      const int m = m0 + r;
      float v = acc[r];
      v = (v > 0.f) ? v : (__expf(v) - 1.f);          // ELU
      C16[(size_t)m * c16ld + colOff + n] = (f16)v;
    }
  } else {
#pragma unroll
    for (int r = 0; r < 8; ++r)
      C32[b * cBatch + (size_t)(m0 + r) * ldc + n] = acc[r];
  }
}

// ---------------------------------------------------------------------------
// Row softmax of leaky_relu(src_i + dst_j) * ea, masked where ea<=0.
// One 256-thread block per (p, i) row; 8 elements/thread in registers.
// Writes the f32 chain (next layer's edge_attr) and an f16 copy for WMMA.
// Safe in-place (ea == eaOut): full row is read before any write.
// ---------------------------------------------------------------------------
__global__ __launch_bounds__(256) void egat_attn(
    const float* __restrict__ ea, const float* __restrict__ esrc,
    const float* __restrict__ edst,
    float* __restrict__ eaOut, f16* __restrict__ att16, int writeEa)
{
  __shared__ float red[256];
  const int i = blockIdx.x;
  const int p = blockIdx.y;
  const size_t base = ((size_t)p * NNODES + i) * NNODES;
  const float si = esrc[i];

  float v[8];
  float mx = -3.0e38f;
#pragma unroll
  for (int t = 0; t < 8; ++t) {
    const int j = threadIdx.x + t * 256;
    const float a = ea[base + j];
    float x = si + edst[j];
    x = (x > 0.f) ? x : 0.2f * x;                     // leaky_relu
    const float m = (a > 0.f) ? (x * a) : -9.0e15f;   // mask
    v[t] = m;
    mx = fmaxf(mx, m);
  }
  red[threadIdx.x] = mx; __syncthreads();
  for (int s = 128; s > 0; s >>= 1) {
    if (threadIdx.x < s) red[threadIdx.x] = fmaxf(red[threadIdx.x], red[threadIdx.x + s]);
    __syncthreads();
  }
  const float rmax = red[0]; __syncthreads();

  float sum = 0.f;
#pragma unroll
  for (int t = 0; t < 8; ++t) { const float e = __expf(v[t] - rmax); v[t] = e; sum += e; }
  red[threadIdx.x] = sum; __syncthreads();
  for (int s = 128; s > 0; s >>= 1) {
    if (threadIdx.x < s) red[threadIdx.x] += red[threadIdx.x + s];
    __syncthreads();
  }
  const float inv = 1.f / red[0];

#pragma unroll
  for (int t = 0; t < 8; ++t) {
    const int j = threadIdx.x + t * 256;
    const float o = v[t] * inv;
    if (writeEa) eaOut[base + j] = o;
    att16[base + j] = (f16)o;
  }
}

// e_src[i] = Wh[i,:].a[:F], e_dst[i] = Wh[i,:].a[F:2F]   (batched over heads)
__global__ void egat_srcdst(const float* __restrict__ Wh, size_t whBatch,
                            const float* __restrict__ a, size_t aBatch,
                            float* __restrict__ esrc, float* __restrict__ edst,
                            size_t eBatch, int Fout)
{
  const int b = blockIdx.y;
  const int i = blockIdx.x * blockDim.x + threadIdx.x;
  const float* w  = Wh + b * whBatch + (size_t)i * Fout;
  const float* av = a + b * aBatch;
  float s1 = 0.f, s2 = 0.f;
  for (int f = 0; f < Fout; ++f) { s1 += w[f] * av[f]; s2 += w[f] * av[Fout + f]; }
  esrc[b * eBatch + i] = s1;
  edst[b * eBatch + i] = s2;
}

// elementwise f32 -> f16 (row-major, for the A operand X)
__global__ void egat_cvt(const float* __restrict__ in, f16* __restrict__ out, size_t n)
{
  const size_t i = (size_t)blockIdx.x * blockDim.x + threadIdx.x;
  if (i < n) out[i] = (f16)in[i];
}

// batched transpose+convert: in[b][k][n] (f32) -> out[b][n][k] (f16, B col-major)
__global__ void egat_tcvt(const float* __restrict__ in, f16* __restrict__ out,
                          int K, int Nn, int batches)
{
  const size_t per   = (size_t)K * Nn;
  const size_t total = per * batches;
  const size_t idx   = (size_t)blockIdx.x * blockDim.x + threadIdx.x;
  if (idx >= total) return;
  const size_t b   = idx / per;
  const size_t rem = idx - b * per;
  const int k = (int)(rem / Nn);
  const int n = (int)(rem - (size_t)k * Nn);
  out[b * per + (size_t)n * K + k] = (f16)in[idx];
}

// mean over P, ELU, log_softmax over NCLASS
__global__ void egat_finalize(const float* __restrict__ hp, float* __restrict__ out)
{
  const int i = blockIdx.x * blockDim.x + threadIdx.x;
  if (i >= NNODES) return;
  float v[NCLASSC];
  float mx = -3.0e38f;
#pragma unroll
  for (int c = 0; c < NCLASSC; ++c) {
    float s = 0.f;
#pragma unroll
    for (int p = 0; p < PP; ++p)
      s += hp[(size_t)p * NNODES * NCLASSC + (size_t)i * NCLASSC + c];
    s *= 0.25f;
    s = (s > 0.f) ? s : (__expf(s) - 1.f);            // ELU
    v[c] = s;
    mx = fmaxf(mx, s);
  }
  float sum = 0.f;
#pragma unroll
  for (int c = 0; c < NCLASSC; ++c) sum += __expf(v[c] - mx);
  const float lse = __logf(sum);
#pragma unroll
  for (int c = 0; c < NCLASSC; ++c) out[(size_t)i * NCLASSC + c] = v[c] - mx - lse;
}

// ---------------------------------------------------------------------------
extern "C" void kernel_launch(void* const* d_in, const int* in_sizes, int n_in,
                              void* d_out, int out_size, void* d_ws, size_t ws_size,
                              hipStream_t stream)
{
  (void)in_sizes; (void)n_in; (void)out_size; (void)ws_size;
  const float* x       = (const float*)d_in[0];   // [2048,256]
  const float* edge    = (const float*)d_in[1];   // [4,2048,2048]
  const float* W_heads = (const float*)d_in[2];   // [4,256,64]
  const float* a_heads = (const float*)d_in[3];   // [4,128,1]
  const float* W_out   = (const float*)d_in[4];   // [1024,16]
  const float* a_out   = (const float*)d_in[5];   // [32,1]

  char* wp = (char*)d_ws;
  auto carve = [&](size_t bytes) -> void* {
    void* r = (void*)wp;
    wp += (bytes + 255) & ~(size_t)255;
    return r;
  };
  float* eaChain  = (float*)carve((size_t)PP * NNODES * NNODES * 4);      // 64 MB
  f16*   att16    = (f16*)  carve((size_t)PP * NNODES * NNODES * 2);      // 32 MB
  f16*   x16      = (f16*)  carve((size_t)NNODES * NFEATC * 2);
  f16*   Whead_cm = (f16*)  carve((size_t)NHEADSC * NHID * NFEATC * 2);
  f16*   Wout_cm  = (f16*)  carve((size_t)NCLASSC * OUT_INC * 2);
  float* Wh32     = (float*)carve((size_t)NHEADSC * NNODES * NHID * 4);
  f16*   Wh16cm   = (f16*)  carve((size_t)NHEADSC * NHID * NNODES * 2);
  float* esrcH    = (float*)carve((size_t)NHEADSC * NNODES * 4);
  float* edstH    = (float*)carve((size_t)NHEADSC * NNODES * 4);
  f16*   xcat16   = (f16*)  carve((size_t)NNODES * OUT_INC * 2);
  float* WhO32    = (float*)carve((size_t)NNODES * NCLASSC * 4);
  f16*   WhO16cm  = (f16*)  carve((size_t)NCLASSC * NNODES * 2);
  float* esrcO    = (float*)carve((size_t)NNODES * 4);
  float* edstO    = (float*)carve((size_t)NNODES * 4);
  float* hp       = (float*)carve((size_t)PP * NNODES * NCLASSC * 4);

  // --- operand precision conversion ---
  egat_cvt <<<dim3((NNODES * NFEATC) / 256), 256, 0, stream>>>(x, x16, (size_t)NNODES * NFEATC);
  egat_tcvt<<<dim3((NHEADSC * NFEATC * NHID) / 256), 256, 0, stream>>>(W_heads, Whead_cm, NFEATC, NHID, NHEADSC);
  egat_tcvt<<<dim3((OUT_INC * NCLASSC) / 256), 256, 0, stream>>>(W_out, Wout_cm, OUT_INC, NCLASSC, 1);

  // --- Wh for all 4 heads (chain-independent): [2048,256]@[256,64], batched ---
  egat_gemm_wmma<<<dim3(128, 1, NHEADSC), 128, 0, stream>>>(
      x16, NFEATC, 0,
      Whead_cm, NFEATC, (size_t)NHID * NFEATC,
      Wh32, NHID, (size_t)NNODES * NHID,
      Wh16cm, NNODES, (size_t)NHID * NNODES,
      0, 0, NNODES, NHID / 16, NFEATC, /*mode=*/0);

  egat_srcdst<<<dim3(NNODES / 256, NHEADSC), 256, 0, stream>>>(
      Wh32, (size_t)NNODES * NHID, a_heads, (size_t)2 * NHID,
      esrcH, edstH, (size_t)NNODES, NHID);

  // --- sequential heads: softmax-attention chain + h_prime GEMM (ELU->xcat) ---
  for (int h = 0; h < NHEADSC; ++h) {
    const float* eaIn = (h == 0) ? edge : eaChain;
    egat_attn<<<dim3(NNODES, PP), 256, 0, stream>>>(
        eaIn, esrcH + (size_t)h * NNODES, edstH + (size_t)h * NNODES,
        eaChain, att16, /*writeEa=*/1);
    // h_prime[p] = att16[p] @ Wh ; ELU ; scatter into xcat[:, h*256 + p*64 + f]
    egat_gemm_wmma<<<dim3(128, 1, PP), 128, 0, stream>>>(
        att16, NNODES, (size_t)NNODES * NNODES,
        Wh16cm + (size_t)h * NHID * NNODES, NNODES, 0,
        nullptr, 0, 0,
        xcat16, OUT_INC, 0,
        /*colOffBase=*/h * PP * NHID, /*colOffPerBatch=*/NHID,
        NNODES, NHID / 16, NNODES, /*mode=*/1);
  }

  // --- output layer: Wh_out = xcat @ W_out  [2048,1024]@[1024,16] ---
  egat_gemm_wmma<<<dim3(32, 1, 1), 128, 0, stream>>>(
      xcat16, OUT_INC, 0,
      Wout_cm, OUT_INC, 0,
      WhO32, NCLASSC, 0,
      WhO16cm, NNODES, 0,
      0, 0, NNODES, NCLASSC / 16, OUT_INC, /*mode=*/0);

  egat_srcdst<<<dim3(NNODES / 256, 1), 256, 0, stream>>>(
      WhO32, 0, a_out, 0, esrcO, edstO, 0, NCLASSC);

  egat_attn<<<dim3(NNODES, PP), 256, 0, stream>>>(
      eaChain, esrcO, edstO, nullptr, att16, /*writeEa=*/0);

  // hp[p] = att16[p] @ Wh_out   [2048,2048]@[2048,16]
  egat_gemm_wmma<<<dim3(32, 1, PP), 128, 0, stream>>>(
      att16, NNODES, (size_t)NNODES * NNODES,
      WhO16cm, NNODES, 0,
      hp, NCLASSC, (size_t)NNODES * NCLASSC,
      nullptr, 0, 0, 0, 0,
      NNODES, NCLASSC / 16, NNODES, /*mode=*/2);

  egat_finalize<<<dim3(NNODES / 256), 256, 0, stream>>>(hp, (float*)d_out);
}
